// EdgeConv_83966610636802
// MI455X (gfx1250) — matrix operand order repaired and checked by reading, not compiled, not run on hardware
//
#include <hip/hip_runtime.h>
#include <hip/hip_bf16.h>
#include <stdint.h>

// Problem constants (match reference)
#define Bb 8
#define Mm 8192
#define Ee 2048
#define Dd 64
#define DT 128            // concatenated width: [edge_scores | edge_features]
#define KSTEPS (Ee / 32)  // 64 K-tiles of 32

typedef __attribute__((ext_vector_type(16))) __bf16 v16bf;
typedef __attribute__((ext_vector_type(8)))  float  v8f;
typedef __attribute__((ext_vector_type(4)))  float  f4;

static __device__ __forceinline__ uint16_t f2bf(float f) {
    uint32_t u = __float_as_uint(f);
    uint32_t r = u + 0x7FFFu + ((u >> 16) & 1u);   // round-to-nearest-even
    return (uint16_t)(r >> 16);
}

// Truncating pack: exact for inc_mat values {0.0, 1.0}. Single v_perm_b32.
static __device__ __forceinline__ uint32_t pack_bf_trunc(float lo, float hi) {
    return __builtin_amdgcn_perm(__float_as_uint(hi), __float_as_uint(lo), 0x07060302u);
}

static __device__ __forceinline__ void sched_fence() {
#if __has_builtin(__builtin_amdgcn_sched_barrier)
    __builtin_amdgcn_sched_barrier(0);   // keep load phase and WMMA phase separate
#endif
}

// ---------------------------------------------------------------------------
// Kernel 1: edge_scores = edge_features @ Wa^T, then emit bf16 concat matrix
// in WMMA-B-friendly layout: edgecat[b][e/32][n][e%32], n in [0,128)
//   n <  64 : edge_scores (attention input)
//   n >= 64 : raw edge_features
// Wa staged in LDS with padded stride (65) to avoid 32-way bank conflicts.
// ---------------------------------------------------------------------------
__global__ void __launch_bounds__(256)
edge_transform_kernel(const float* __restrict__ ef,
                      const float* __restrict__ Wa,
                      uint16_t* __restrict__ edgecat) {
    __shared__ float WaS[Dd * 65];
    __shared__ float xs[4][Dd];
    const int t  = threadIdx.x;
    const int le = t >> 6;      // local edge 0..3
    const int o  = t & 63;      // output feature
    const int b  = blockIdx.y;
    const int e  = blockIdx.x * 4 + le;

#pragma unroll
    for (int i = 0; i < (Dd * Dd) / 256; ++i) {
        int lin = t + i * 256;
        WaS[(lin >> 6) * 65 + (lin & 63)] = Wa[lin];
    }
    const float* xrow = ef + ((size_t)(b * Ee + e)) * Dd;
    xs[le][o] = xrow[o];
    __syncthreads();

    float s = 0.f;
#pragma unroll 8
    for (int k = 0; k < Dd; ++k) s += WaS[o * 65 + k] * xs[le][k];

    const size_t base = (((size_t)b * KSTEPS + (e >> 5)) * DT) * 32 + (size_t)(e & 31);
    edgecat[base + (size_t)o * 32]        = f2bf(s);
    edgecat[base + (size_t)(64 + o) * 32] = f2bf(xs[le][o]);
}

// ---------------------------------------------------------------------------
// Kernel 2: main fused GEMM  C[b] (M x 128) = inc[b] (M x E) * edgecat[b]
//   cols   0..63  -> node_scores (pre-softmax)
//   cols 64..127  -> agg = inc @ edge_features
// 256 threads / block = 8 waves; block tile 128(M) x 128(N), K-step 32.
// B tiles: global_load_async_to_lds_b128 (ASYNCcnt, double buffered)
// A tiles: NT fp32 loads -> v_perm packs (exact for binary data) -> LDS
// Compute: sched_barrier splits 18x ds_load_b128 from 8 back-to-back WMMAs.
// ---------------------------------------------------------------------------
__global__ void __launch_bounds__(256)
incmat_gemm_kernel(const float* __restrict__ inc,
                   const uint16_t* __restrict__ edgecat,
                   float* __restrict__ scores,
                   float* __restrict__ agg) {
    __shared__ uint16_t Ash[2][128 * 32];   // 8 KB x2, row-major bf16
    __shared__ uint16_t Bsh[2][128 * 32];   // 8 KB x2, [n][k] bf16

    const int t    = threadIdx.x;
    const int b    = blockIdx.y;
    const int rb   = blockIdx.x;        // 128-row block
    const int lane = t & 31;
    const int wave = t >> 5;            // 8 waves: wave w -> rows [16w,16w+16)

    // A staging mapping: thread t covers row lr, k-half kh
    const int lr = t >> 1;              // 0..127
    const int kh = (t & 1) * 16;        // 0 or 16

    const float* Abase = inc + ((size_t)(b * Mm + rb * 128 + lr)) * Ee + kh;

    const uint64_t BgAddr   = (uint64_t)(uintptr_t)(edgecat + ((size_t)b * KSTEPS) * (DT * 32));
    const uint32_t BldsB[2] = { (uint32_t)(uintptr_t)&Bsh[0][0],
                                (uint32_t)(uintptr_t)&Bsh[1][0] };

    v8f acc[8];
#pragma unroll
    for (int j = 0; j < 8; ++j) acc[j] = (v8f){0, 0, 0, 0, 0, 0, 0, 0};

    auto stageA = [&](int kt, int buf) {
        const f4* g = (const f4*)(Abase + (size_t)kt * 32);
        f4 f0 = __builtin_nontemporal_load(g + 0);   // stream inc_mat: keep L2 for B/out
        f4 f1 = __builtin_nontemporal_load(g + 1);
        f4 f2 = __builtin_nontemporal_load(g + 2);
        f4 f3 = __builtin_nontemporal_load(g + 3);
        uint32_t* d = (uint32_t*)&Ash[buf][lr * 32 + kh];
        *(uint4*)(d + 0) = make_uint4(pack_bf_trunc(f0.x, f0.y), pack_bf_trunc(f0.z, f0.w),
                                      pack_bf_trunc(f1.x, f1.y), pack_bf_trunc(f1.z, f1.w));
        *(uint4*)(d + 4) = make_uint4(pack_bf_trunc(f2.x, f2.y), pack_bf_trunc(f2.z, f2.w),
                                      pack_bf_trunc(f3.x, f3.y), pack_bf_trunc(f3.z, f3.w));
    };

    auto issueB = [&](int kt, int buf) {
        // 8 KB tile: 256 threads x 32 B (two async b128 per thread)
        uint32_t lds  = BldsB[buf] + (uint32_t)t * 32;
        uint32_t goff = (uint32_t)kt * (DT * 32 * 2) + (uint32_t)t * 32;
        asm volatile("global_load_async_to_lds_b128 %0, %1, %2 offset:0"
                     :: "v"(lds), "v"(goff), "s"(BgAddr) : "memory");
        asm volatile("global_load_async_to_lds_b128 %0, %1, %2 offset:16"
                     :: "v"(lds), "v"(goff), "s"(BgAddr) : "memory");
    };

    union Frag { uint4 q[2]; v16bf v; };

    auto compute = [&](int buf) {
        // A fragment (bf16 16x32): row m=lane%16 of this wave's band
        Frag af;
        const uint16_t* ap = &Ash[buf][(wave * 16 + (lane & 15)) * 32 + (lane >> 4) * 16];
        af.q[0] = *(const uint4*)(ap);        // K +0..7
        af.q[1] = *(const uint4*)(ap + 8);    // K +8..15

        // All 8 B fragments (bf16 32x16) into distinct registers
        Frag bfr[8];
#pragma unroll
        for (int j = 0; j < 8; ++j) {
            const uint16_t* bp = &Bsh[buf][(j * 16 + (lane & 15)) * 32 + (lane >> 4) * 16];
            bfr[j].q[0] = *(const uint4*)(bp);       // K +0..7
            bfr[j].q[1] = *(const uint4*)(bp + 8);   // K +8..15
        }
        sched_fence();   // loads stay above, WMMAs below: one dscnt wait, 8 XDL issues
#pragma unroll
        for (int j = 0; j < 8; ++j) {
            acc[j] = __builtin_amdgcn_wmma_f32_16x16x32_bf16(
                false, af.v, false, bfr[j].v, (short)0, acc[j], false, false);
        }
    };

    // Prime buffer 0
    stageA(0, 0);
    issueB(0, 0);

    for (int kt = 0; kt < KSTEPS; ++kt) {
        const int cur = kt & 1, nxt = cur ^ 1;
        __syncthreads();                       // nxt buffer free for reuse
        if (kt + 1 < KSTEPS) {
            issueB(kt + 1, nxt);
            stageA(kt + 1, nxt);
            asm volatile("s_wait_asynccnt 0x2" ::: "memory");  // cur's copies done
        } else {
            asm volatile("s_wait_asynccnt 0x0" ::: "memory");
        }
        __syncthreads();                       // all waves' cur copies visible
        compute(cur);
    }

    // Write out: C/D layout -> VGPR r holds (M = 16*wave + r + 8*(lane/16), N = lane%16)
#pragma unroll
    for (int j = 0; j < 8; ++j) {
        const int col = j * 16 + (lane & 15);
        float* dst = (col < 64) ? scores : agg;
        const int c = col & 63;
#pragma unroll
        for (int r = 0; r < 8; ++r) {
            const int m = rb * 128 + wave * 16 + r + 8 * (lane >> 4);
            dst[((size_t)(b * Mm + m)) * Dd + c] = acc[j][r];
        }
    }
}

// ---------------------------------------------------------------------------
// Kernel 3: softmax statistics over the M axis (axis=1) per (b, d):
// max and sum(exp(x-max)) over 8192 rows. scores was just written through L2,
// so the strided column reads hit L2.
// ---------------------------------------------------------------------------
__global__ void __launch_bounds__(256)
softmax_stats_kernel(const float* __restrict__ scores,
                     float* __restrict__ stats) {
    __shared__ float red[256];
    const int d = blockIdx.x, b = blockIdx.y, t = threadIdx.x;
    const float* col = scores + ((size_t)b * Mm) * Dd + d;

    float vals[Mm / 256];
    float mx = -3.4e38f;
#pragma unroll 4
    for (int i = 0; i < Mm / 256; ++i) {
        float v = col[(size_t)(t + i * 256) * Dd];
        vals[i] = v;
        mx = fmaxf(mx, v);
    }
    red[t] = mx;
    __syncthreads();
    for (int s = 128; s > 0; s >>= 1) {
        if (t < s) red[t] = fmaxf(red[t], red[t + s]);
        __syncthreads();
    }
    mx = red[0];
    __syncthreads();

    float sum = 0.f;
#pragma unroll 4
    for (int i = 0; i < Mm / 256; ++i) sum += __expf(vals[i] - mx);
    red[t] = sum;
    __syncthreads();
    for (int s = 128; s > 0; s >>= 1) {
        if (t < s) red[t] += red[t + s];
        __syncthreads();
    }
    if (t == 0) {
        stats[(b * Dd + d) * 2 + 0] = mx;
        stats[(b * Dd + d) * 2 + 1] = red[0];
    }
}

// ---------------------------------------------------------------------------
// Kernel 4: weighted = agg * softmax(scores); out = a*prev + (1-a)*(weighted@Wp^T)
// Wp staged in LDS (padded stride 65); prev/out use non-temporal traffic.
// ---------------------------------------------------------------------------
__global__ void __launch_bounds__(256)
finalize_kernel(const float* __restrict__ scores,
                const float* __restrict__ agg,
                const float* __restrict__ stats,
                const float* __restrict__ Wp,
                const float* __restrict__ prev,
                const float* __restrict__ alphaPtr,
                float* __restrict__ out) {
    __shared__ float WpS[Dd * 65];
    __shared__ float ws[4][Dd];
    const int t  = threadIdx.x;
    const int lr = t >> 6, o = t & 63;
    const int b  = blockIdx.y;
    const int m  = blockIdx.x * 4 + lr;
    const size_t ridx = ((size_t)(b * Mm + m)) * Dd;

#pragma unroll
    for (int i = 0; i < (Dd * Dd) / 256; ++i) {
        int lin = t + i * 256;
        WpS[(lin >> 6) * 65 + (lin & 63)] = Wp[lin];
    }

    const float mx = stats[(b * Dd + o) * 2 + 0];
    const float sm = stats[(b * Dd + o) * 2 + 1];
    const float attn = __expf(scores[ridx + o] - mx) / sm;
    ws[lr][o] = agg[ridx + o] * attn;
    __syncthreads();

    float nf = 0.f;
#pragma unroll 8
    for (int k = 0; k < Dd; ++k) nf += WpS[o * 65 + k] * ws[lr][k];

    const float alpha = alphaPtr[0];
    const float pv = __builtin_nontemporal_load(prev + ridx + o);
    __builtin_nontemporal_store(alpha * pv + (1.f - alpha) * nf, out + ridx + o);
}

// ---------------------------------------------------------------------------
extern "C" void kernel_launch(void* const* d_in, const int* in_sizes, int n_in,
                              void* d_out, int out_size, void* d_ws, size_t ws_size,
                              hipStream_t stream) {
    const float* inc   = (const float*)d_in[0];
    const float* ef    = (const float*)d_in[1];
    const float* prev  = (const float*)d_in[2];
    const float* Wa    = (const float*)d_in[3];
    const float* Wp    = (const float*)d_in[4];
    const float* alpha = (const float*)d_in[5];

    char* ws = (char*)d_ws;
    uint16_t* edgecat = (uint16_t*)(ws);                      // 4 MB  bf16 [b][e/32][n][e%32]
    float*    scores  = (float*)(ws + (size_t)(4  << 20));    // 16 MB (B,M,64)
    float*    aggbuf  = (float*)(ws + (size_t)(20 << 20));    // 16 MB (B,M,64)
    float*    stats   = (float*)(ws + (size_t)(36 << 20));    // 4 KB  (B,64,2)
    float*    out     = (float*)d_out;

    edge_transform_kernel<<<dim3(Ee / 4, Bb), 256, 0, stream>>>(ef, Wa, edgecat);
    incmat_gemm_kernel   <<<dim3(Mm / 128, Bb), 256, 0, stream>>>(inc, edgecat, scores, aggbuf);
    softmax_stats_kernel <<<dim3(Dd, Bb), 256, 0, stream>>>(scores, stats);
    finalize_kernel      <<<dim3(Mm / 4, Bb), 256, 0, stream>>>(scores, aggbuf, stats, Wp, prev, alpha, out);
}
